// MultilayerGRU_91182155694661
// MI455X (gfx1250) — matrix operand order repaired and checked by reading, not compiled, not run on hardware
//
#include <hip/hip_runtime.h>
#include <cstdint>
#include <cstddef>

// ---------------------------------------------------------------------------
// 2-layer GRU for MI455X (gfx1250, wave32, WMMA bf16 16x16x32, fp32 accum).
//  - persistent kernel, software-pipelined layers (layer0 step s || layer1 step s-1)
//  - monotonic-counter grid barrier between pipeline steps
//  - weights pre-packed into WMMA-B fragment layout (bf16)
//  - batched output GEMM + final-hidden writeback
// ---------------------------------------------------------------------------

#define DEVINL __device__ __forceinline__

constexpr int Bsz = 32;     // batch
constexpr int Seq = 1024;   // sequence length
constexpr int Idim = 256;   // input dim (layer0 x-side K)
constexpr int Hdim = 1024;  // hidden dim
constexpr int Odim = 256;   // output dim
constexpr int NWG = 32;     // persistent workgroups (well below WGP count -> co-resident)

typedef __attribute__((ext_vector_type(16))) __bf16 v16bf;
typedef __attribute__((ext_vector_type(8)))  __bf16 v8bf;
typedef __attribute__((ext_vector_type(8)))  float  v8f;

union V16U { v16bf v; v8bf h[2]; };

DEVINL __bf16 f2bf(float f) {
  unsigned u = __builtin_bit_cast(unsigned, f);
  unsigned r = (u + 0x7FFFu + ((u >> 16) & 1u)) >> 16;   // RNE
  unsigned short s = (unsigned short)r;
  return __builtin_bit_cast(__bf16, s);
}
DEVINL float bf2f(__bf16 b) {
  unsigned short s = __builtin_bit_cast(unsigned short, b);
  unsigned u = ((unsigned)s) << 16;
  return __builtin_bit_cast(float, u);
}
DEVINL float sigmoidf_(float x) { return 1.0f / (1.0f + __expf(-x)); }
DEVINL v8f zero8() { v8f z = {0.f,0.f,0.f,0.f,0.f,0.f,0.f,0.f}; return z; }

DEVINL v8f wmma_bf16(v16bf a, v16bf b, v8f c) {
  // D = A(16x32 bf16) * B(32x16 bf16) + C(16x16 f32)
  return __builtin_amdgcn_wmma_f32_16x16x32_bf16(
      /*neg_a=*/false, a, /*neg_b=*/false, b,
      /*c_mod=*/(short)0, c, /*reuse_a=*/false, /*reuse_b=*/false);
}

// A fragment (16x32, 16-bit): lane = half*16 + M; per-lane halves:
//   i in 0..7  -> K = kb*32 + 8*half + i
//   i in 8..15 -> K = kb*32 + 16 + 8*half + (i-8)
// Row-major A with stride `rowStride` (elements) gives two contiguous 16B loads.
DEVINL v16bf load_a_frag(const __bf16* __restrict__ base, int rowStride,
                         int m0, int kb, int lane) {
  int m = m0 + (lane & 15);
  int half = lane >> 4;
  const __bf16* p = base + (size_t)m * rowStride + kb * 32 + 8 * half;
  V16U r;
  r.h[0] = *(const v8bf*)(p);
  r.h[1] = *(const v8bf*)(p + 16);
  return r.v;
}

// B fragment: pre-packed so each lane reads 32 contiguous bytes.
DEVINL v16bf load_b_frag(const __bf16* __restrict__ packed, int kblocks,
                         int panel, int kb, int lane) {
  const __bf16* p = packed + (((size_t)(panel * kblocks + kb) * 32 + lane) << 4);
  V16U r;
  r.h[0] = *(const v8bf*)(p);
  r.h[1] = *(const v8bf*)(p + 8);
  return r.v;
}

// ---------------------------------------------------------------------------
// Pre-pass kernels
// ---------------------------------------------------------------------------

// Pack W (N x K row-major fp32, torch Linear weight) into bf16 WMMA-B fragments.
// B(k,n) = W[n][k]. Fragment element: lane = half*16 + (n%16), i<8 -> K = kb*32 + 8*half + i, i>=8 -> K = kb*32 + 16 + 8*half + (i-8).
__global__ void pack_weight(const float* __restrict__ W, __bf16* __restrict__ dst,
                            int K, int kblocks) {
  size_t t = (size_t)blockIdx.x * blockDim.x + threadIdx.x;
  int i    = (int)(t & 15);
  int lane = (int)((t >> 4) & 31);
  int kb   = (int)((t >> 9) % (size_t)kblocks);
  int panel= (int)(t / ((size_t)512 * kblocks));
  int n = panel * 16 + (lane & 15);
  int k = kb * 32 + 8 * (lane >> 4) + (i < 8 ? i : 8 + i);
  dst[t] = f2bf(W[(size_t)n * K + k]);
}

__global__ void convert_x(const float* __restrict__ x, __bf16* __restrict__ xb, size_t n) {
  size_t t = (size_t)blockIdx.x * blockDim.x + threadIdx.x;
  if (t < n) xb[t] = f2bf(x[t]);
}

// Init recurrent state (bf16) from hidden_state (B, L, H) fp32; zero sync counter.
__global__ void init_state(const float* __restrict__ hidden,
                           __bf16* __restrict__ h0r, __bf16* __restrict__ h1r,
                           unsigned* cnt) {
  int t = blockIdx.x * blockDim.x + threadIdx.x;
  if (t == 0) *cnt = 0u;
  if (t < Bsz * Hdim) {
    int b = t / Hdim, k = t % Hdim;
    h0r[t] = f2bf(hidden[(size_t)b * 2 * Hdim + k]);
    h1r[t] = f2bf(hidden[(size_t)b * 2 * Hdim + Hdim + k]);
  }
}

// ---------------------------------------------------------------------------
// Persistent recurrent kernel
// ---------------------------------------------------------------------------

struct RecArgs {
  const __bf16* xbf;                 // (B, S, I) bf16
  __bf16* h0a; __bf16* h0b;          // layer0 state, double buffered (B, H)
  __bf16* h1a; __bf16* h1b;          // layer1 state, double buffered (B, H)
  __bf16* h1seq;                     // (B, S, H) layer1 outputs
  const __bf16* pw0[6];              // layer0 packed: xz, xr, xg, hz, hr, hg
  const __bf16* pw1[6];              // layer1 packed
  const float* bz0; const float* br0; const float* bg0;
  const float* bz1; const float* br1; const float* bg1;
  unsigned* cnt;                     // grid-barrier monotonic counter
};

__global__ __launch_bounds__(256, 1) void gru_persistent(RecArgs P) {
  const int waveId = blockIdx.x * (blockDim.x >> 5) + (threadIdx.x >> 5); // 0..255
  const int layer  = waveId >> 7;        // 0 or 1
  const int tile   = waveId & 127;       // 0..127
  const int m0     = (tile & 1) * 16;    // batch-row half
  const int panel  = tile >> 1;          // 0..63 -> 16 output columns each
  const int n0     = panel * 16;
  const int lane   = threadIdx.x & 31;

  const __bf16* const* pw = layer ? P.pw1 : P.pw0;
  const float* bz = layer ? P.bz1 : P.bz0;
  const float* br = layer ? P.br1 : P.br0;
  const float* bg = layer ? P.bg1 : P.bg0;

  #pragma unroll 1
  for (int s = 0; s <= Seq; ++s) {
    const bool active = layer ? (s >= 1) : (s < Seq);
    if (active) {
      const int t  = layer ? (s - 1) : s;
      const int rd = (s + 1) & 1;        // state written at step s-1
      const int wr = s & 1;

      const __bf16* Ax; int axStride, Kxb;
      if (layer == 0) { Ax = P.xbf + (size_t)t * Idim; axStride = Seq * Idim; Kxb = Idim / 32; }
      else            { Ax = rd ? P.h0b : P.h0a;       axStride = Hdim;       Kxb = Hdim / 32; }
      const __bf16* Ah = layer ? (rd ? P.h1b : P.h1a) : (rd ? P.h0b : P.h0a);
      __bf16*       Hw = layer ? (wr ? P.h1b : P.h1a) : (wr ? P.h0b : P.h0a);

      v8f az = zero8(), ar = zero8(), agx = zero8(), agh = zero8();

      // x-side: z, r accumulate; g x-part kept separate (r gates only the h-part)
      #pragma unroll 2
      for (int kb = 0; kb < Kxb; ++kb) {
        v16bf a = load_a_frag(Ax, axStride, m0, kb, lane);
        az  = wmma_bf16(a, load_b_frag(pw[0], Kxb, panel, kb, lane), az);
        ar  = wmma_bf16(a, load_b_frag(pw[1], Kxb, panel, kb, lane), ar);
        agx = wmma_bf16(a, load_b_frag(pw[2], Kxb, panel, kb, lane), agx);
      }
      // h-side
      constexpr int Khb = Hdim / 32;
      #pragma unroll 2
      for (int kb = 0; kb < Khb; ++kb) {
        v16bf a = load_a_frag(Ah, Hdim, m0, kb, lane);
        az  = wmma_bf16(a, load_b_frag(pw[3], Khb, panel, kb, lane), az);
        ar  = wmma_bf16(a, load_b_frag(pw[4], Khb, panel, kb, lane), ar);
        agh = wmma_bf16(a, load_b_frag(pw[5], Khb, panel, kb, lane), agh);
      }

      // Fused gate epilogue. D layout: reg r, lane l -> row = m0 + r + 8*(l/16), col = n0 + l%16
      const int col  = n0 + (lane & 15);
      const int half = lane >> 4;
      const float bzv = bz[col], brv = br[col], bgv = bg[col];
      #pragma unroll
      for (int r = 0; r < 8; ++r) {
        int b = m0 + r + 8 * half;                         // batch index
        float hp = bf2f(Ah[(size_t)b * Hdim + col]);       // h_prev
        float z  = sigmoidf_(az[r] + bzv);
        float rr = sigmoidf_(ar[r] + brv);
        float g  = tanhf(agx[r] + rr * (agh[r] + bgv));
        float hn = z * hp + (1.0f - z) * g;
        __bf16 hb = f2bf(hn);
        Hw[(size_t)b * Hdim + col] = hb;
        if (layer) P.h1seq[((size_t)b * Seq + t) * Hdim + col] = hb;
      }
    }

    // ---- grid barrier (monotonic counter; 32 WGs are co-resident) ----
    __syncthreads();
    if (threadIdx.x == 0) {
      __threadfence();                       // release our state writes
      atomicAdd(P.cnt, 1u);
      const unsigned target = (unsigned)(s + 1) * NWG;
      while (__atomic_load_n(P.cnt, __ATOMIC_RELAXED) < target)
        __builtin_amdgcn_s_sleep(8);
      __threadfence();                       // acquire other WGs' writes
    }
    __syncthreads();
  }
}

// ---------------------------------------------------------------------------
// Batched output projection: y = h1seq @ W_hy^T + b_hy   (M=B*S, N=O, K=H)
// ---------------------------------------------------------------------------
__global__ void out_gemm(const __bf16* __restrict__ hseq, const __bf16* __restrict__ pwhy,
                         const float* __restrict__ bias, float* __restrict__ out) {
  int wave = blockIdx.x * (blockDim.x >> 5) + (threadIdx.x >> 5); // 0..32767
  int m0 = (wave >> 4) * 16;   // 2048 M-tiles
  int np = wave & 15;          // 16 N-panels
  int n0 = np * 16;
  int lane = threadIdx.x & 31;
  v8f acc = zero8();
  constexpr int Kb = Hdim / 32;
  #pragma unroll 4
  for (int kb = 0; kb < Kb; ++kb) {
    v16bf a = load_a_frag(hseq, Hdim, m0, kb, lane);
    v16bf b = load_b_frag(pwhy, Kb, np, kb, lane);
    acc = wmma_bf16(a, b, acc);
  }
  int col = n0 + (lane & 15), half = lane >> 4;
  float bv = bias[col];
  #pragma unroll
  for (int r = 0; r < 8; ++r) {
    int row = m0 + r + 8 * half;                // row = b*S + t, matches (B,S,O) flat
    out[(size_t)row * Odim + col] = acc[r] + bv;
  }
}

// Final hidden state (B, L, H) fp32 at tail of d_out.
__global__ void write_hfinal(const __bf16* __restrict__ h0f, const __bf16* __restrict__ h1f,
                             float* __restrict__ dst) {
  int t = blockIdx.x * blockDim.x + threadIdx.x;
  if (t < Bsz * 2 * Hdim) {
    int b = t / (2 * Hdim);
    int rem = t % (2 * Hdim);
    int l = rem / Hdim, k = rem % Hdim;
    const __bf16* src = l ? h1f : h0f;
    dst[t] = bf2f(src[(size_t)b * Hdim + k]);
  }
}

// ---------------------------------------------------------------------------
// Host launch
// ---------------------------------------------------------------------------
extern "C" void kernel_launch(void* const* d_in, const int* in_sizes, int n_in,
                              void* d_out, int out_size, void* d_ws, size_t ws_size,
                              hipStream_t stream) {
  (void)in_sizes; (void)n_in; (void)out_size; (void)ws_size;
  char* ws = (char*)d_ws;
  size_t off = 0;
  auto take = [&](size_t bytes) -> void* {
    void* p = ws + off;
    off += (bytes + 255) & ~(size_t)255;
    return p;
  };

  unsigned* cnt = (unsigned*)take(256);
  __bf16* h0[2] = { (__bf16*)take((size_t)Bsz * Hdim * 2), (__bf16*)take((size_t)Bsz * Hdim * 2) };
  __bf16* h1[2] = { (__bf16*)take((size_t)Bsz * Hdim * 2), (__bf16*)take((size_t)Bsz * Hdim * 2) };
  __bf16* xbf   = (__bf16*)take((size_t)Bsz * Seq * Idim * 2);
  __bf16* h1seq = (__bf16*)take((size_t)Bsz * Seq * Hdim * 2);

  // d_in order: x, hidden_state, then per layer {w_xz,w_hz,b_hz,w_xr,w_hr,b_hr,w_xg,w_hg,b_hg}, w_hy, b_hy
  struct WD { int src; int K; int N; };
  const WD wd[13] = {
    {2, Idim, Hdim}, {5, Idim, Hdim}, {8, Idim, Hdim},     // l0: xz, xr, xg
    {3, Hdim, Hdim}, {6, Hdim, Hdim}, {9, Hdim, Hdim},     // l0: hz, hr, hg
    {11, Hdim, Hdim}, {14, Hdim, Hdim}, {17, Hdim, Hdim},  // l1: xz, xr, xg
    {12, Hdim, Hdim}, {15, Hdim, Hdim}, {18, Hdim, Hdim},  // l1: hz, hr, hg
    {20, Hdim, Odim}                                       // w_hy
  };
  __bf16* pw[13];
  for (int i = 0; i < 13; ++i) {
    size_t elems = (size_t)wd[i].K * wd[i].N;
    pw[i] = (__bf16*)take(elems * 2);
    pack_weight<<<(unsigned)(elems / 256), 256, 0, stream>>>(
        (const float*)d_in[wd[i].src], pw[i], wd[i].K, wd[i].K / 32);
  }

  init_state<<<(Bsz * Hdim) / 256, 256, 0, stream>>>(
      (const float*)d_in[1], h0[1], h1[0], cnt);

  size_t nx = (size_t)Bsz * Seq * Idim;
  convert_x<<<(unsigned)(nx / 256), 256, 0, stream>>>((const float*)d_in[0], xbf, nx);

  RecArgs P;
  P.xbf = xbf;
  P.h0a = h0[0]; P.h0b = h0[1]; P.h1a = h1[0]; P.h1b = h1[1];
  P.h1seq = h1seq;
  for (int g = 0; g < 6; ++g) { P.pw0[g] = pw[g]; P.pw1[g] = pw[6 + g]; }
  P.bz0 = (const float*)d_in[4];  P.br0 = (const float*)d_in[7];  P.bg0 = (const float*)d_in[10];
  P.bz1 = (const float*)d_in[13]; P.br1 = (const float*)d_in[16]; P.bg1 = (const float*)d_in[19];
  P.cnt = cnt;
  gru_persistent<<<NWG, 256, 0, stream>>>(P);

  // y = h1seq @ W_hy^T + b_hy : 32768 wave-tiles, 8 waves per block
  out_gemm<<<((Bsz * Seq / 16) * (Odim / 16)) / 8, 256, 0, stream>>>(
      h1seq, pw[12], (const float*)d_in[21], (float*)d_out);

  write_hfinal<<<(Bsz * 2 * Hdim) / 256, 256, 0, stream>>>(
      h0[1], h1[0], (float*)d_out + (size_t)Bsz * Seq * Odim);
}
